// NeighborWiseAttention_54125177864975
// MI455X (gfx1250) — compile-verified, matrix operand order checked by reading, not run on hardware
//
#include <hip/hip_runtime.h>
#include <hip/hip_bf16.h>

// NeighborWiseAttention for MI455X (gfx1250), wave32 + WMMA bf16 + TDM.
// One 256-thread block (8 waves) per batch row; each wave owns a 16-row block
// of the 128-padded neighbor dimension across all stages.

#define NNEI    120
#define NPAD    128
#define ED      64
#define HD      128
#define NLAYER  2
#define SCALING 0.08838834764831845f   // HD^-0.5
#define LN_EPS  1e-5f

typedef __attribute__((ext_vector_type(16))) __bf16        v16bf;
typedef __attribute__((ext_vector_type(8)))  float         v8f;
typedef __attribute__((ext_vector_type(4)))  unsigned int  v4u;
typedef __attribute__((ext_vector_type(4)))  unsigned int  u32x4;
typedef __attribute__((ext_vector_type(4)))  int           i32x4;
typedef __attribute__((ext_vector_type(8)))  int           i32x8;

union FragAB { v16bf v; v4u q[2]; unsigned short h[16]; };
union FragC  { v8f   v; float f[8]; };

// ---- LDS layout (bytes); kernel has no static LDS, so dynamic LDS starts at 0
#define OFF_X    0u        // float  [128][64]  32768  current x / residual
#define OFF_R    32768u    // float  [128][3]    1536  unit r-vectors
#define OFF_M    34304u    // float  [128]        512  neighbor mask (0/1)
#define OFF_B    34816u    // float  [384]       1536  in_b for current layer
#define OFF_LN   36352u    // float  [128]        512  ln_g | ln_b
#define OFF_A    36864u    // ushort [128][128] 32768  q  -> aw -> o
#define OFF_K    69632u    // ushort [128][128] 32768  k  -> out_w^T
#define OFF_VT   102400u   // ushort [128][128] 32768  v^T
#define OFF_W    135168u   // ushort [128][32]   8192  in_w slice (transposed)
#define SMEM_BYTES 143360u

#if __has_builtin(__builtin_amdgcn_tensor_load_to_lds) && \
    __has_builtin(__builtin_amdgcn_tensor_store_from_lds) && \
    __has_builtin(__builtin_amdgcn_s_wait_tensorcnt)
#define USE_TDM 1
#else
#define USE_TDM 0
#endif

__device__ __forceinline__ unsigned short f2bf(float x) {
  union { float f; unsigned int u; } c; c.f = x;
  unsigned int u = c.u;
  u += 0x7fffu + ((u >> 16) & 1u);   // round-to-nearest-even
  return (unsigned short)(u >> 16);
}

// Reductions across the 16 lanes of a wave32 half (xor masks < 16 stay in-half).
__device__ __forceinline__ float hsum16(float x) {
  x += __shfl_xor(x, 1, 32);
  x += __shfl_xor(x, 2, 32);
  x += __shfl_xor(x, 4, 32);
  x += __shfl_xor(x, 8, 32);
  return x;
}
__device__ __forceinline__ float hmax16(float x) {
  x = fmaxf(x, __shfl_xor(x, 1, 32));
  x = fmaxf(x, __shfl_xor(x, 2, 32));
  x = fmaxf(x, __shfl_xor(x, 4, 32));
  x = fmaxf(x, __shfl_xor(x, 8, 32));
  return x;
}

__device__ __forceinline__ v8f wmma_bf16(const FragAB& a, const FragAB& b, v8f c) {
  return __builtin_amdgcn_wmma_f32_16x16x32_bf16(false, a.v, false, b.v,
                                                 (short)0, c, false, false);
}

#if USE_TDM
// Build a 2D Tensor-DMA descriptor (D#) per CDNA5 ISA 8.3/8.4 and issue
// TENSOR_LOAD_TO_LDS / TENSOR_STORE_FROM_LDS. data_size fixed at 4 bytes.
// dim0 = row length (f32 elems, == tile width), dim1 = #rows, stride0 in elems.
__device__ __forceinline__ void tdm_copy_2d(int store, unsigned lds_addr,
                                            const void* gptr,
                                            unsigned dim0, unsigned dim1,
                                            unsigned stride0)
{
  unsigned long long ga = (unsigned long long)(size_t)gptr;
  u32x4 g0;
  g0[0] = 1u;                                            // count=1, user mode
  g0[1] = lds_addr;                                      // lds_addr[31:0]
  g0[2] = (unsigned)(ga & 0xffffffffu);                  // global_addr[31:0]
  g0[3] = (unsigned)((ga >> 32) & 0x1ffffffu) | (2u << 30); // addr[56:32] | type=2
  i32x8 g1;
  g1[0] = (int)(2u << 16);                               // data_size = 4B (code 2)
  g1[1] = (int)((dim0 & 0xffffu) << 16);                 // tensor_dim0[15:0]
  g1[2] = (int)(((dim0 >> 16) & 0xffffu) | ((dim1 & 0xffffu) << 16));
  g1[3] = (int)(((dim1 >> 16) & 0xffffu) | ((dim0 & 0xffffu) << 16)); // tile_dim0
  g1[4] = (int)(dim1 & 0xffffu);                         // tile_dim1, tile_dim2=0
  g1[5] = (int)stride0;                                  // tensor_dim0_stride[31:0]
  g1[6] = 0;
  g1[7] = 0;
  i32x4 z4 = {0, 0, 0, 0};
#if __clang_major__ >= 23
  i32x8 z8 = {0, 0, 0, 0, 0, 0, 0, 0};
  if (store) __builtin_amdgcn_tensor_store_from_lds(g0, g1, z4, z4, z8, 0);
  else       __builtin_amdgcn_tensor_load_to_lds  (g0, g1, z4, z4, z8, 0);
#else
  if (store) __builtin_amdgcn_tensor_store_from_lds(g0, g1, z4, z4, 0);
  else       __builtin_amdgcn_tensor_load_to_lds  (g0, g1, z4, z4, 0);
#endif
}
#endif // USE_TDM

__global__ void __launch_bounds__(256)
nwattn_kernel(const float* __restrict__ G, const unsigned char* __restrict__ nmask,
              const float* __restrict__ rin, const float* __restrict__ in_w,
              const float* __restrict__ in_b, const float* __restrict__ out_w,
              const float* __restrict__ out_b, const float* __restrict__ ln_g,
              const float* __restrict__ ln_b, float* __restrict__ out)
{
  extern __shared__ __align__(16) char smem[];
  float*          sX  = (float*)(smem + OFF_X);
  float*          sR  = (float*)(smem + OFF_R);
  float*          sM  = (float*)(smem + OFF_M);
  float*          sB  = (float*)(smem + OFF_B);
  float*          sLN = (float*)(smem + OFF_LN);
  unsigned short* sA  = (unsigned short*)(smem + OFF_A);
  unsigned short* sK  = (unsigned short*)(smem + OFF_K);
  unsigned short* sVT = (unsigned short*)(smem + OFF_VT);
  unsigned short* sW  = (unsigned short*)(smem + OFF_W);

  const int b    = blockIdx.x;
  const int tid  = threadIdx.x;
  const int lane = tid & 31;
  const int wave = tid >> 5;
  const int half = lane >> 4;     // WMMA half-wave (rows 0-7 vs 8-15 of C/D)
  const int ln   = lane & 15;     // N / column lane within tile
  const int R    = wave << 4;     // this wave's 16-row block base

  // ---------------- stage 0: per-row inputs -> LDS ----------------
  const float*         gX = G    + (size_t)b * (NNEI * ED);
  const float*         gR = rin  + (size_t)b * (NNEI * 3);
  const unsigned char* gM = nmask + (size_t)b * NNEI;

#if USE_TDM
  if (wave == 0) {
    // x tile: 120 rows x 64 f32, contiguous; r: 360 f32 as a 1-row tile.
    tdm_copy_2d(0, OFF_X, gX, ED, NNEI, ED);
    tdm_copy_2d(0, OFF_R, gR, NNEI * 3, 1, NNEI * 3);
  }
  for (int i = NNEI * ED + tid; i < NPAD * ED; i += 256) sX[i] = 0.0f; // pad rows
  for (int i = NNEI * 3 + tid; i < NPAD * 3; i += 256) sR[i] = 0.0f;
#else
  for (int i = tid; i < NPAD * ED; i += 256)
    sX[i] = (i < NNEI * ED) ? gX[i] : 0.0f;
  for (int i = tid; i < NPAD * 3; i += 256)
    sR[i] = (i < NNEI * 3) ? gR[i] : 0.0f;
#endif
  for (int i = tid; i < NPAD; i += 256)
    sM[i] = (i < NNEI && gM[i]) ? 1.0f : 0.0f;
  __builtin_prefetch(in_w, 0, 0);
  __builtin_prefetch(out_w, 0, 0);
#if USE_TDM
  if (wave == 0) __builtin_amdgcn_s_wait_tensorcnt(0);
#endif
  __syncthreads();

  for (int l = 0; l < NLAYER; ++l) {
    const float* wL  = in_w  + (size_t)l * ED * (3 * HD);   // [64][384]
    const float* bL  = in_b  + l * (3 * HD);
    const float* owL = out_w + l * HD * ED;                  // [128][64]
    const float* obL = out_b + l * ED;
    for (int i = tid; i < 3 * HD; i += 256) sB[i] = bL[i];
    for (int i = tid; i < ED; i += 256) {
      sLN[i]      = ln_g[l * ED + i];
      sLN[ED + i] = ln_b[l * ED + i];
    }
    // first barrier inside the ks loop below publishes sB/sLN

    // ---------------- stage 1: qkv = x @ in_w + b, l2norm, store ----------------
    for (int s = 0; s < 3; ++s) {          // 0=q 1=k 2=v (128 cols each)
      FragC acc[8];
      #pragma unroll
      for (int j = 0; j < 8; ++j)
        #pragma unroll
        for (int i = 0; i < 8; ++i) acc[j].f[i] = 0.0f;

      for (int ks = 0; ks < 2; ++ks) {     // K = 64 = 2 x 32
        __syncthreads();                   // previous sW readers done
        // stage in_w slice transposed: sW[f][kk] = wL[ks*32+kk][s*128+f]
        for (int i = tid; i < 128 * 32; i += 256) {
          int kk = i >> 7, f = i & 127;
          sW[f * 32 + kk] = f2bf(wL[(ks * 32 + kk) * (3 * HD) + s * 128 + f]);
        }
        __syncthreads();

        // A fragment: rows of x (f32 -> bf16 in-register)
        FragAB af;
        const float* xrow = sX + (R + ln) * ED + ks * 32 + 8 * half;
        #pragma unroll
        for (int e = 0; e < 8; ++e) { af.h[e] = f2bf(xrow[e]); af.h[8 + e] = f2bf(xrow[16 + e]); }

        // preload all B fragments, then run the WMMA chain back-to-back
        FragAB bfr[8];
        #pragma unroll
        for (int j = 0; j < 8; ++j) {
          const unsigned short* wrow = sW + (j * 16 + ln) * 32 + 8 * half;
          bfr[j].q[0] = *(const v4u*)(wrow);
          bfr[j].q[1] = *(const v4u*)(wrow + 16);
        }
        #pragma unroll
        for (int j = 0; j < 8; ++j)
          acc[j].v = wmma_bf16(af, bfr[j], acc[j].v);
      }

      // +bias, row l2-norm over HD=128, store bf16 (q row-major, k row-major, v transposed)
      const int base = s * 128;
      #pragma unroll
      for (int v = 0; v < 8; ++v) {
        #pragma unroll
        for (int j = 0; j < 8; ++j) acc[j].f[v] += sB[base + j * 16 + ln];
        float ss = 0.0f;
        #pragma unroll
        for (int j = 0; j < 8; ++j) ss += acc[j].f[v] * acc[j].f[v];
        ss = hsum16(ss);
        float scale = 1.0f / fmaxf(sqrtf(ss), 1e-12f);
        if (s == 0) scale *= SCALING;
        const int row = R + v + 8 * half;
        if (row >= NNEI) scale = 0.0f;     // zero padded rows
        #pragma unroll
        for (int j = 0; j < 8; ++j) {
          unsigned short hb = f2bf(acc[j].f[v] * scale);
          const int h = j * 16 + ln;
          if      (s == 0) sA[row * HD + h]    = hb;   // q[n][h]
          else if (s == 1) sK[row * HD + h]    = hb;   // k[m][h]
          else             sVT[h * NPAD + row] = hb;   // v^T[h][m]
        }
      }
    }
    __syncthreads();   // q/k/v^T visible to all waves

    // ---------------- stage 2: aw = q @ k^T, mask, softmax, *ang ----------------
    {
      FragC acc[8];
      #pragma unroll
      for (int j = 0; j < 8; ++j)
        #pragma unroll
        for (int i = 0; i < 8; ++i) acc[j].f[i] = 0.0f;

      #pragma unroll
      for (int ks = 0; ks < 4; ++ks) {     // K = HD = 128
        FragAB af;
        const unsigned short* qrow = sA + (R + ln) * HD + ks * 32 + 8 * half;
        af.q[0] = *(const v4u*)(qrow);
        af.q[1] = *(const v4u*)(qrow + 16);
        FragAB bfr[8];
        #pragma unroll
        for (int j = 0; j < 8; ++j) {
          const unsigned short* krow = sK + (j * 16 + ln) * HD + ks * 32 + 8 * half;
          bfr[j].q[0] = *(const v4u*)(krow);
          bfr[j].q[1] = *(const v4u*)(krow + 16);
        }
        #pragma unroll
        for (int j = 0; j < 8; ++j)
          acc[j].v = wmma_bf16(af, bfr[j], acc[j].v);
      }
      __syncthreads();   // q/k reads done -> sA reusable for aw, sK reusable later

      const float NEG_INF = -__builtin_inff();
      #pragma unroll
      for (int v = 0; v < 8; ++v) {
        const int row = R + v + 8 * half;
        float mx = NEG_INF;
        #pragma unroll
        for (int j = 0; j < 8; ++j) {
          float a = acc[j].f[v];
          a = (sM[j * 16 + ln] > 0.5f) ? a : NEG_INF;  // mask_k (+padded cols)
          acc[j].f[v] = a;
          mx = fmaxf(mx, a);
        }
        mx = hmax16(mx);
        if (mx == NEG_INF) mx = 0.0f;                  // fully-masked row guard
        float sum = 0.0f;
        #pragma unroll
        for (int j = 0; j < 8; ++j) {
          float p = __expf(acc[j].f[v] - mx);
          acc[j].f[v] = p;
          sum += p;
        }
        sum = hsum16(sum);
        const float rowm = sM[row] / fmaxf(sum, 1e-30f);   // softmax norm * mask_q
        const float rx = sR[row * 3], ry = sR[row * 3 + 1], rz = sR[row * 3 + 2];
        #pragma unroll
        for (int j = 0; j < 8; ++j) {
          const int m = j * 16 + ln;
          const float ang = rx * sR[m * 3] + ry * sR[m * 3 + 1] + rz * sR[m * 3 + 2];
          sA[row * NPAD + m] = f2bf(acc[j].f[v] * rowm * ang);  // own rows only
        }
      }
    }

    // stage out_w^T into sK region (sK readers finished at the barrier above)
    for (int i = tid; i < HD * ED; i += 256) {
      int h = i >> 6, e = i & 63;
      sK[e * HD + h] = f2bf(owL[i]);
    }

    // ---------------- stage 3a: o = aw @ v ----------------
    {
      FragC oacc[8];
      #pragma unroll
      for (int j = 0; j < 8; ++j)
        #pragma unroll
        for (int i = 0; i < 8; ++i) oacc[j].f[i] = 0.0f;

      #pragma unroll
      for (int ks = 0; ks < 4; ++ks) {     // K = NPAD = 128
        FragAB af;
        const unsigned short* arow = sA + (R + ln) * NPAD + ks * 32 + 8 * half;
        af.q[0] = *(const v4u*)(arow);
        af.q[1] = *(const v4u*)(arow + 16);
        FragAB bfr[8];
        #pragma unroll
        for (int j = 0; j < 8; ++j) {
          const unsigned short* vrow = sVT + (j * 16 + ln) * NPAD + ks * 32 + 8 * half;
          bfr[j].q[0] = *(const v4u*)(vrow);
          bfr[j].q[1] = *(const v4u*)(vrow + 16);
        }
        #pragma unroll
        for (int j = 0; j < 8; ++j)
          oacc[j].v = wmma_bf16(af, bfr[j], oacc[j].v);
      }
      // overwrite own aw rows with o (bf16); same-wave LDS ops are in-order
      #pragma unroll
      for (int v = 0; v < 8; ++v) {
        const int row = R + v + 8 * half;
        #pragma unroll
        for (int j = 0; j < 8; ++j)
          sA[row * HD + j * 16 + ln] = f2bf(oacc[j].f[v]);
      }
    }
    __syncthreads();   // out_w^T staged + o stored everywhere

    // ---------------- stage 3b: x = LN(residual + o @ out_w + out_b) ----------------
    {
      FragC pacc[4];
      #pragma unroll
      for (int j = 0; j < 4; ++j)
        #pragma unroll
        for (int i = 0; i < 8; ++i) pacc[j].f[i] = 0.0f;

      #pragma unroll
      for (int ks = 0; ks < 4; ++ks) {     // K = HD = 128
        FragAB af;
        const unsigned short* orow = sA + (R + ln) * HD + ks * 32 + 8 * half;
        af.q[0] = *(const v4u*)(orow);
        af.q[1] = *(const v4u*)(orow + 16);
        FragAB bfr[4];
        #pragma unroll
        for (int j = 0; j < 4; ++j) {
          const unsigned short* wrow = sK + (j * 16 + ln) * HD + ks * 32 + 8 * half;
          bfr[j].q[0] = *(const v4u*)(wrow);
          bfr[j].q[1] = *(const v4u*)(wrow + 16);
        }
        #pragma unroll
        for (int j = 0; j < 4; ++j)
          pacc[j].v = wmma_bf16(af, bfr[j], pacc[j].v);
      }

      #pragma unroll
      for (int v = 0; v < 8; ++v) {
        const int row = R + v + 8 * half;
        float xv[4];
        float mu = 0.0f;
        #pragma unroll
        for (int j = 0; j < 4; ++j) {
          const int e = j * 16 + ln;
          float x = pacc[j].f[v] + sX[row * ED + e] + obL[e];
          xv[j] = x;
          mu += x;
        }
        mu = hsum16(mu) * (1.0f / 64.0f);
        float var = 0.0f;
        #pragma unroll
        for (int j = 0; j < 4; ++j) { float d = xv[j] - mu; var += d * d; }
        var = hsum16(var) * (1.0f / 64.0f);
        const float rst = rsqrtf(var + LN_EPS);
        #pragma unroll
        for (int j = 0; j < 4; ++j) {
          const int e = j * 16 + ln;
          sX[row * ED + e] = (xv[j] - mu) * rst * sLN[e] + sLN[ED + e];
        }
      }
    }
    __syncthreads();   // end of layer: sLN/sB/sK/sVT safe to rewrite
  }

  // ---------------- write result ----------------
  float* gOut = out + (size_t)b * (NNEI * ED);
#if USE_TDM
  if (wave == 0) {
    tdm_copy_2d(1, OFF_X, gOut, ED, NNEI, ED);
    __builtin_amdgcn_s_wait_tensorcnt(0);
  }
#else
  for (int i = tid; i < NNEI * ED; i += 256) gOut[i] = sX[i];
#endif
}

extern "C" void kernel_launch(void* const* d_in, const int* in_sizes, int n_in,
                              void* d_out, int out_size, void* d_ws, size_t ws_size,
                              hipStream_t stream) {
  (void)n_in; (void)d_ws; (void)ws_size; (void)out_size;
  const float*         G     = (const float*)d_in[0];
  const unsigned char* nmask = (const unsigned char*)d_in[1];  // jnp bool_ = 1 byte
  const float*         rin   = (const float*)d_in[2];
  const float*         in_w  = (const float*)d_in[3];
  const float*         in_b  = (const float*)d_in[4];
  const float*         out_w = (const float*)d_in[5];
  const float*         out_b = (const float*)d_in[6];
  const float*         ln_g  = (const float*)d_in[7];
  const float*         ln_b  = (const float*)d_in[8];
  float*               out   = (float*)d_out;

  const int batch = in_sizes[0] / (NNEI * ED);   // 4096

  (void)hipFuncSetAttribute((const void*)nwattn_kernel,
                            hipFuncAttributeMaxDynamicSharedMemorySize,
                            (int)SMEM_BYTES);
  nwattn_kernel<<<batch, 256, SMEM_BYTES, stream>>>(
      G, nmask, rin, in_w, in_b, out_w, out_b, ln_g, ln_b, out);
}
